// ContinuousSSMLayer_86131274154306
// MI455X (gfx1250) — matrix operand order repaired and checked by reading, not compiled, not run on hardware
//
#include <hip/hip_runtime.h>

// ---------------------------------------------------------------------------
// ContinuousSSMLayer for MI455X (gfx1250, wave32, WMMA).
// Pipeline: setup -> WMMA GEMM (Bu = u @ B_bar^T) -> 3-pass chunked scan
//           -> WMMA GEMM (y = 2Re(C~ x) + D u).
// GEMMs in bf16 WMMA w/ f32 accumulate (memory-bound at 23.3 TB/s);
// recurrence in exact fp32. Weight tiles staged into LDS with the gfx1250
// async global->LDS path (ASYNCcnt) when the toolchain exposes it.
// ---------------------------------------------------------------------------

typedef __attribute__((ext_vector_type(16))) __bf16 v16bf;
typedef __attribute__((ext_vector_type(8)))  float  v8f;
typedef int v4i __attribute__((vector_size(16)));   // matches builtin proto "V4i"

#if __has_builtin(__builtin_amdgcn_global_load_async_to_lds_b128) && \
    __has_builtin(__builtin_amdgcn_s_wait_asynccnt)
#define USE_ASYNC_LDS 1
#else
#define USE_ASYNC_LDS 0
#endif

#define BATCH 16
#define SEQL  8192
#define HDIM  256
#define PDIM  256
#define MTOT  (BATCH * SEQL)      // 131072 rows
#define LCH   128                 // scan chunk length
#define NCH   (SEQL / LCH)        // 64 chunks

// workspace layout (bytes)
#define OFF_W    ((size_t)0)                     // Wt  : [512][256] bf16 (256 KB)
#define OFF_C2   ((size_t)(512 * 256 * 2))       // C2t : [256][512] bf16 (256 KB)
#define OFF_ABRE ((size_t)(OFF_C2 + 256 * 512 * 2))
#define OFF_ABIM (OFF_ABRE + 1024)
#define OFF_QRE  (OFF_ABIM + 1024)               // Lambda_bar^LCH
#define OFF_QIM  (OFF_QRE + 1024)
#define OFF_ERE  ((size_t)(1u << 20))            // chunk end states [B][NCH][P]
#define OFF_EIM  ((size_t)(2u << 20))
#define OFF_CRE  ((size_t)(3u << 20))            // chunk carries
#define OFF_CIM  ((size_t)(4u << 20))
#define OFF_BU   ((size_t)(8u << 20))            // Bu/xs [M][512] f32 (256 MB)

static __device__ inline unsigned short f2bf(float f) {
    unsigned int u = __builtin_bit_cast(unsigned int, f);
    u = u + 0x7FFFu + ((u >> 16) & 1u);          // round-to-nearest-even
    return (unsigned short)(u >> 16);
}

struct Bits32B { uint4 lo, hi; };
static __device__ inline v16bf make_frag(uint4 lo, uint4 hi) {
    Bits32B t{lo, hi};
    return __builtin_bit_cast(v16bf, t);
}

// ---------------------------------------------------------------------------
// Setup: ZOH discretization + bf16 weight packing.
//   Wt [n][k]  n<256: Re(B_bar[p=n][h=k]);  n>=256: Im(B_bar[p=n-256][h=k])
//   C2t[h][k]  k<256: 2*Re(C[h][p=k]);      k>=256: -2*Im(C[h][p=k-256])
// ---------------------------------------------------------------------------
__global__ void ssm_setup(const float* __restrict__ Lre, const float* __restrict__ Lim,
                          const float* __restrict__ Bm,  const float* __restrict__ Cm,
                          const float* __restrict__ log_step,
                          unsigned short* __restrict__ Wt, unsigned short* __restrict__ C2t,
                          float* __restrict__ ab_re, float* __restrict__ ab_im,
                          float* __restrict__ q_re,  float* __restrict__ q_im) {
    const int t = threadIdx.x;                           // one block, 256 threads
    // ---- per-mode scalars (t == p) ----
    const float dt = __expf(log_step[t]);
    const float lr = Lre[t], li = Lim[t];
    const float e  = __expf(lr * dt);
    const float abr = e * __cosf(li * dt);
    const float abi = e * __sinf(li * dt);
    ab_re[t] = abr; ab_im[t] = abi;
    float pr = abr, pi = abi;                            // Lambda_bar^128 by squaring
    for (int s = 0; s < 7; ++s) { float nr = pr*pr - pi*pi; float ni = 2.f*pr*pi; pr = nr; pi = ni; }
    q_re[t] = pr; q_im[t] = pi;
    // ---- g = (Lambda_bar - 1) / Lambda ----
    const float den = lr*lr + li*li;
    const float gr = ((abr - 1.f)*lr + abi*li) / den;
    const float gi = (abi*lr - (abr - 1.f)*li) / den;
    // ---- B_bar rows (t == p) ----
    for (int h = 0; h < HDIM; ++h) {
        const float b0 = Bm[t*2*HDIM + 2*h + 0];
        const float b1 = Bm[t*2*HDIM + 2*h + 1];
        Wt[t*HDIM + h]          = f2bf(gr*b0 - gi*b1);   // Re
        Wt[(PDIM + t)*HDIM + h] = f2bf(gr*b1 + gi*b0);   // Im
    }
    // ---- C rows (t == h) ----
    for (int p = 0; p < PDIM; ++p) {
        const float c0 = Cm[t*2*PDIM + 2*p + 0];
        const float c1 = Cm[t*2*PDIM + 2*p + 1];
        C2t[t*512 + p]        = f2bf( 2.f * c0);
        C2t[t*512 + PDIM + p] = f2bf(-2.f * c1);
    }
}

// ---------------------------------------------------------------------------
// WMMA GEMM: Out[M x NLD] = A[M x K] (f32->bf16) * Bg[N x K]^T (bf16).
// Block tile 128x128, 8 waves of 32x64, K staged in 32-wide LDS tiles,
// double-buffered; B tile staged via async global->LDS when available.
// LDS rows padded to 40 bf16 (20 dwords) -> conflict-free ds_load_b128.
// EPI: Out += Dvec[col] * U[row*NLD+col] (output projection epilogue).
// ---------------------------------------------------------------------------
template <int K, int NLD, bool EPI>
__global__ __launch_bounds__(256)
void ssm_gemm128(const float* __restrict__ A, const unsigned short* __restrict__ Bg,
                 float* __restrict__ Out, const float* __restrict__ Dvec,
                 const float* __restrict__ U) {
    __shared__ __align__(16) unsigned short As[2][128 * 40];
    __shared__ __align__(16) unsigned short Bs[2][128 * 40];

    const int tid  = threadIdx.x;
    const int lane = tid & 31;
    const int wave = tid >> 5;
    const int wm   = wave & 3;          // 4 wave-rows of 32
    const int wn   = wave >> 2;         // 2 wave-cols of 64
    const int blockM = blockIdx.y * 128;
    const int blockN = blockIdx.x * 128;
    const int NS = K / 32;

    // stage K-step s into LDS buffer `buf`
    auto stage = [&](int s, int buf) {
        const int ks = s * 32;
        // A tile (128x32 f32 -> bf16): 4 float4 per thread
        #pragma unroll
        for (int j = 0; j < 4; ++j) {
            const int idx = j * 256 + tid;
            const int row = idx >> 3;
            const int q   = idx & 7;
            const float4 v = *(const float4*)(A + (size_t)(blockM + row) * K + ks + q * 4);
            const unsigned int u0 = (unsigned int)f2bf(v.x) | ((unsigned int)f2bf(v.y) << 16);
            const unsigned int u1 = (unsigned int)f2bf(v.z) | ((unsigned int)f2bf(v.w) << 16);
            *(uint2*)(&As[buf][row * 40 + q * 4]) = make_uint2(u0, u1);
        }
        // prefetch next stage's A addresses into cache (global_prefetch_b8)
        if (ks + 32 < K)
            __builtin_prefetch(A + (size_t)(blockM + (tid >> 3)) * K + ks + 32 + (tid & 7) * 4, 0, 1);
        // B tile (128x32 bf16): 2 x b128 per thread
        #pragma unroll
        for (int j = 0; j < 2; ++j) {
            const int idx = j * 256 + tid;
            const int row = idx >> 2;
            const int q   = idx & 3;
            const unsigned short* gp = Bg + (size_t)(blockN + row) * K + ks + q * 8;
            unsigned short* lp = &Bs[buf][row * 40 + q * 8];
#if USE_ASYNC_LDS
            __builtin_amdgcn_global_load_async_to_lds_b128(
                (__attribute__((address_space(1))) v4i*)gp,
                (__attribute__((address_space(3))) v4i*)lp, 0, 0);
#else
            *(uint4*)lp = *(const uint4*)gp;
#endif
        }
    };

    v8f acc[2][4];
    const v8f vzero = {0.f, 0.f, 0.f, 0.f, 0.f, 0.f, 0.f, 0.f};
    for (int mi = 0; mi < 2; ++mi)
        for (int ni = 0; ni < 4; ++ni) acc[mi][ni] = vzero;

    stage(0, 0);
    for (int s = 0; s < NS; ++s) {
        const int buf = s & 1;
#if USE_ASYNC_LDS
        __builtin_amdgcn_s_wait_asynccnt(0);   // own async B-tile loads landed in LDS
#endif
        __syncthreads();                       // all waves' A stores + async loads visible
        if (s + 1 < NS) stage(s + 1, buf ^ 1); // overlap next-stage DMA with WMMA below

        // fragments (ISA 7.12.2): A lane m=lane%16, k = 8*(lane/16)+{0..7, 16..23}
        //                         B lane n=lane%16, k = 8*(lane/16)+{0..7, 16..23}
        v16bf af[2], bfm[4];
        const int ko = (lane >> 4) << 3;
        #pragma unroll
        for (int mi = 0; mi < 2; ++mi) {
            const int r = wm * 32 + mi * 16 + (lane & 15);
            af[mi] = make_frag(*(const uint4*)(&As[buf][r * 40 + ko]),
                               *(const uint4*)(&As[buf][r * 40 + ko + 16]));
        }
        const int kb = (lane >> 4) << 3;
        #pragma unroll
        for (int ni = 0; ni < 4; ++ni) {
            const int n = wn * 64 + ni * 16 + (lane & 15);
            bfm[ni] = make_frag(*(const uint4*)(&Bs[buf][n * 40 + kb]),
                                *(const uint4*)(&Bs[buf][n * 40 + kb + 16]));
        }
        #pragma unroll
        for (int mi = 0; mi < 2; ++mi)
            #pragma unroll
            for (int ni = 0; ni < 4; ++ni)
                acc[mi][ni] = __builtin_amdgcn_wmma_f32_16x16x32_bf16(
                    false, af[mi], false, bfm[ni], (short)0, acc[mi][ni], false, false);
    }

    // epilogue (C layout: vgpr r -> M = 8*(lane/16)+r, N = lane%16)
    #pragma unroll
    for (int mi = 0; mi < 2; ++mi) {
        const int rowbase = blockM + wm * 32 + mi * 16 + ((lane >> 4) << 3);
        #pragma unroll
        for (int ni = 0; ni < 4; ++ni) {
            const int gcol = blockN + wn * 64 + ni * 16 + (lane & 15);
            if constexpr (EPI) {
                const float dv = Dvec[gcol];
                #pragma unroll
                for (int r = 0; r < 8; ++r) {
                    const int grow = rowbase + r;
                    Out[(size_t)grow * NLD + gcol] =
                        acc[mi][ni][r] + dv * U[(size_t)grow * NLD + gcol];
                }
            } else {
                #pragma unroll
                for (int r = 0; r < 8; ++r) {
                    const int grow = rowbase + r;
                    __builtin_nontemporal_store(acc[mi][ni][r], &Out[(size_t)grow * NLD + gcol]);
                }
            }
        }
    }
}

// ---------------------------------------------------------------------------
// Chunked scan, x[l] = Lambda_bar * x[l-1] + Bu[l], constant per-mode decay.
// Lanes cover p -> fully coalesced 1KB row segments of Bu.
// ---------------------------------------------------------------------------
__global__ __launch_bounds__(256)
void scan_pass1(const float* __restrict__ Bu, const float* __restrict__ ab_re,
                const float* __restrict__ ab_im, float* __restrict__ E_re,
                float* __restrict__ E_im) {
    const int b = blockIdx.x >> 6, c = blockIdx.x & 63, p = threadIdx.x;
    const float ar = ab_re[p], ai = ab_im[p];
    float xr = 0.f, xi = 0.f;
    const size_t base = ((size_t)(b * SEQL + c * LCH)) * 512;
    #pragma unroll 4
    for (int i = 0; i < LCH; ++i) {
        const size_t row = base + (size_t)i * 512;
        const float br = __builtin_nontemporal_load(&Bu[row + p]);
        const float bi = __builtin_nontemporal_load(&Bu[row + PDIM + p]);
        const float nr = ar * xr - ai * xi + br;
        const float ni = ar * xi + ai * xr + bi;
        xr = nr; xi = ni;
    }
    E_re[blockIdx.x * 256 + p] = xr;
    E_im[blockIdx.x * 256 + p] = xi;
}

__global__ __launch_bounds__(256)
void scan_pass2(const float* __restrict__ E_re, const float* __restrict__ E_im,
                const float* __restrict__ q_re, const float* __restrict__ q_im,
                float* __restrict__ C_re, float* __restrict__ C_im) {
    const int b = blockIdx.x, p = threadIdx.x;
    const float qr = q_re[p], qi = q_im[p];
    float xr = 0.f, xi = 0.f;
    for (int c = 0; c < NCH; ++c) {
        const int idx = (b * NCH + c) * 256 + p;
        C_re[idx] = xr; C_im[idx] = xi;               // carry INTO chunk c
        const float er = E_re[idx], ei = E_im[idx];
        const float nr = qr * xr - qi * xi + er;
        const float ni = qr * xi + qi * xr + ei;
        xr = nr; xi = ni;
    }
}

__global__ __launch_bounds__(256)
void scan_pass3(float* __restrict__ Bu, const float* __restrict__ ab_re,
                const float* __restrict__ ab_im, const float* __restrict__ C_re,
                const float* __restrict__ C_im) {
    const int b = blockIdx.x >> 6, c = blockIdx.x & 63, p = threadIdx.x;
    const float ar = ab_re[p], ai = ab_im[p];
    float xr = C_re[blockIdx.x * 256 + p];
    float xi = C_im[blockIdx.x * 256 + p];
    const size_t base = ((size_t)(b * SEQL + c * LCH)) * 512;
    #pragma unroll 4
    for (int i = 0; i < LCH; ++i) {
        const size_t row = base + (size_t)i * 512;
        const float br = Bu[row + p];
        const float bi = Bu[row + PDIM + p];
        const float nr = ar * xr - ai * xi + br;
        const float ni = ar * xi + ai * xr + bi;
        xr = nr; xi = ni;
        Bu[row + p]        = xr;                      // xs, in place
        Bu[row + PDIM + p] = xi;
    }
}

// ---------------------------------------------------------------------------
extern "C" void kernel_launch(void* const* d_in, const int* in_sizes, int n_in,
                              void* d_out, int out_size, void* d_ws, size_t ws_size,
                              hipStream_t stream) {
    (void)in_sizes; (void)n_in; (void)out_size; (void)ws_size;
    const float* u        = (const float*)d_in[0];
    const float* Lre      = (const float*)d_in[1];
    const float* Lim      = (const float*)d_in[2];
    const float* Bm       = (const float*)d_in[3];
    const float* Cm       = (const float*)d_in[4];
    const float* Dv       = (const float*)d_in[5];
    const float* log_step = (const float*)d_in[6];
    float* out = (float*)d_out;
    char*  ws  = (char*)d_ws;

    unsigned short* Wt  = (unsigned short*)(ws + OFF_W);
    unsigned short* C2t = (unsigned short*)(ws + OFF_C2);
    float* ab_re = (float*)(ws + OFF_ABRE);
    float* ab_im = (float*)(ws + OFF_ABIM);
    float* q_re  = (float*)(ws + OFF_QRE);
    float* q_im  = (float*)(ws + OFF_QIM);
    float* E_re  = (float*)(ws + OFF_ERE);
    float* E_im  = (float*)(ws + OFF_EIM);
    float* C_re  = (float*)(ws + OFF_CRE);
    float* C_im  = (float*)(ws + OFF_CIM);
    float* Bu    = (float*)(ws + OFF_BU);

    ssm_setup<<<1, 256, 0, stream>>>(Lre, Lim, Bm, Cm, log_step, Wt, C2t,
                                     ab_re, ab_im, q_re, q_im);
    // Bu[m][n] = u @ B_bar^T  (M=131072, N=512, K=256)
    ssm_gemm128<HDIM, 512, false><<<dim3(4, MTOT / 128), 256, 0, stream>>>(
        u, Wt, Bu, nullptr, nullptr);
    // chunked recurrence (in-place Bu -> xs)
    scan_pass1<<<BATCH * NCH, 256, 0, stream>>>(Bu, ab_re, ab_im, E_re, E_im);
    scan_pass2<<<BATCH, 256, 0, stream>>>(E_re, E_im, q_re, q_im, C_re, C_im);
    scan_pass3<<<BATCH * NCH, 256, 0, stream>>>(Bu, ab_re, ab_im, C_re, C_im);
    // y = 2 Re(C~ x) + D u  (M=131072, N=256, K=512)
    ssm_gemm128<512, HDIM, true><<<dim3(2, MTOT / 128), 256, 0, stream>>>(
        Bu, C2t, out, Dv, u);
}
